// BiLSTM_CRF_52450140618898
// MI455X (gfx1250) — matrix operand order repaired and checked by reading, not compiled, not run on hardware
//
#include <hip/hip_runtime.h>
#include <stdint.h>

// Problem constants (match reference)
#define SLEN  512
#define BATCH 128
#define EMB   256
#define HID   256
#define GATES 1024   // 4*HID, gate order i,f,g,o
#define HD2   512    // concat hidden
#define TAGS  50
#define TPAD  64     // padded tagset for WMMA N-tiling

typedef __attribute__((ext_vector_type(16))) __bf16 bf16x16;
typedef __attribute__((ext_vector_type(8)))  __bf16 bf16x8;
typedef __attribute__((ext_vector_type(8)))  float  f32x8;

static __device__ __forceinline__ f32x8 wmma_bf16(bf16x16 a, bf16x16 b, f32x8 c) {
  // D = A(16x32 bf16) * B(32x16 bf16) + C(16x16 f32)
  return __builtin_amdgcn_wmma_f32_16x16x32_bf16(false, a, false, b, (short)0, c,
                                                 false, false);
}

// A fragment (16x32 bf16, wave32): lanes 0-15 row=lane hold K=k0+{0..7,16..23},
// lanes 16-31 row=lane-16 hold K=k0+{8..15,24..31}. Two 16B loads per lane.
static __device__ __forceinline__ bf16x16 load_a_frag(const __bf16* row, int k0, int base) {
  bf16x8 lo = *(const bf16x8*)(row + k0 + base);
  bf16x8 hi = *(const bf16x8*)(row + k0 + 16 + base);
  return __builtin_shufflevector(lo, hi, 0, 1, 2, 3, 4, 5, 6, 7,
                                 8, 9, 10, 11, 12, 13, 14, 15);
}

static __device__ __forceinline__ float sigf(float x) {
  return 1.0f / (1.0f + __expf(-x));
}

// ---------------------------------------------------------------------------
// K0a: pre-swizzle a weight matrix W[N,K] (row-major, computing X@W^T) into
// WMMA B-fragment order: tile (kt,nt) of 32x16 stored as 32 lanes x 16 bf16
// contiguous (one 32B load per lane at GEMM time).
// B layout: lanes 0-15 hold K=kt*32+{0..7,16..23}, lanes 16-31 K=kt*32+{8..15,24..31}.
// ---------------------------------------------------------------------------
__global__ void k_swizzle_w(const float* __restrict__ W, int Nreal, int Npad, int K,
                            __bf16* __restrict__ out) {
  int total = Npad * K;
  int id = blockIdx.x * blockDim.x + threadIdx.x;
  if (id >= total) return;
  int KT = K >> 5;
  int tile = id >> 9, in = id & 511;
  int lane = in >> 4, j = in & 15;
  int kt = tile % KT, nt = tile / KT;
  int k = (kt << 5) + ((j >> 3) << 4) + ((lane >> 4) << 3) + (j & 7);
  int n = (nt << 4) + (lane & 15);
  float v = (n < Nreal) ? W[(size_t)n * K + k] : 0.0f;
  out[id] = (__bf16)v;
}

// K0b: fold LSTM biases (b_ih + b_hh) per direction.
__global__ void k_bias(const float* __restrict__ bif, const float* __restrict__ bhf,
                       const float* __restrict__ bib, const float* __restrict__ bhb,
                       float* __restrict__ bF, float* __restrict__ bB) {
  int id = blockIdx.x * blockDim.x + threadIdx.x;
  if (id < GATES) bF[id] = bif[id] + bhf[id];
  else if (id < 2 * GATES) { int i = id - GATES; bB[i] = bib[i] + bhb[i]; }
}

// K1: embedding gather -> bf16 activations xs[s*B+b][e]
__global__ void k_embed(const int* __restrict__ sent, const float* __restrict__ emb,
                        __bf16* __restrict__ xs) {
  int sb = blockIdx.x;              // s*BATCH + b
  int s = sb >> 7, b = sb & 127;
  int tok = sent[b * SLEN + s];
  xs[(size_t)sb * EMB + threadIdx.x] = (__bf16)emb[(size_t)tok * EMB + threadIdx.x];
}

// ---------------------------------------------------------------------------
// K2: fused BiLSTM recurrence. grid.x = 16: blocks 0-7 forward (batch tiles),
// 8-15 backward. 256 threads = 8 waves; wave w owns hidden cols [32w,32w+32)
// and computes all 4 gates for its slice -> gate fusion stays in registers.
// x@W_ih fused per step (saves ~1 GB of gate-buffer HBM traffic).
// An opaque zero offset (SGPR asm) inside the time loop defeats LICM of the
// loop-invariant B fragments (no scratch spills) while keeping the bases in
// inferred-global address space -> global_load_b128 saddr from L2-resident
// weights, tracked by LOADcnt only (no flat/DScnt interference with the
// ds_load_b128 A-fragment traffic).
// ---------------------------------------------------------------------------
__global__ void __launch_bounds__(256) k_lstm(
    const __bf16* __restrict__ xs,
    const __bf16* __restrict__ wih_f, const __bf16* __restrict__ wih_b,
    const __bf16* __restrict__ whh_f, const __bf16* __restrict__ whh_b,
    const float* __restrict__ bias_f, const float* __restrict__ bias_b,
    __bf16* __restrict__ h_all) {
  const int XP = 264;                       // row pitch (bf16) -> 528B, 16B aligned
  __shared__ __bf16 x_lds[16 * XP];
  __shared__ __bf16 h_lds[16 * XP];

  int tid = threadIdx.x, wave = tid >> 5, lane = tid & 31;
  int dir = blockIdx.x >> 3, bt = blockIdx.x & 7;
  int b0 = bt << 4;
  const __bf16* wih0 = dir ? wih_b : wih_f;   // select of kernel args: stays
  const __bf16* whh0 = dir ? whh_b : whh_f;   // addrspace(1) after inference
  const float* bias = dir ? bias_b : bias_f;

  for (int i = tid; i < 16 * XP; i += 256) h_lds[i] = (__bf16)0.0f;

  int nlo = lane & 15;
  int base = (lane >> 4) << 3;              // A-frag K sub-offset (0 or 8)
  int mrow = (lane >> 4) << 3;              // C-frag row offset (0 or 8)

  float bI[2], bF[2], bG[2], bO[2];
  for (int ct = 0; ct < 2; ++ct) {
    int col = (wave << 5) + (ct << 4) + nlo;
    bI[ct] = bias[col];
    bF[ct] = bias[256 + col];
    bG[ct] = bias[512 + col];
    bO[ct] = bias[768 + col];
  }
  f32x8 c_reg[2] = {};                      // cell state lives in VGPRs

  int xrow = tid >> 4, xcol = (tid & 15) << 4;
  const __bf16* arow_x = &x_lds[nlo * XP];
  const __bf16* arow_h = &h_lds[nlo * XP];

  for (int t = 0; t < SLEN; ++t) {
    int s = dir ? (SLEN - 1 - t) : t;
    __syncthreads();                        // prev-iter LDS traffic done
    {                                       // cooperative x-tile load (32B/thread)
      const __bf16* src = xs + ((size_t)(s * BATCH + b0 + xrow) * EMB + xcol);
      bf16x8 v0 = *(const bf16x8*)src;
      bf16x8 v1 = *(const bf16x8*)(src + 8);
      *(bf16x8*)&x_lds[xrow * XP + xcol] = v0;
      *(bf16x8*)&x_lds[xrow * XP + xcol + 8] = v1;
    }
    __syncthreads();

    // Opaque zero offsets: values the optimizer cannot prove loop-invariant,
    // so the weight loads stay inside the recurrence (no spill), while the
    // pointer bases keep their global address space (saddr global loads).
    int ofs_w = 0, ofs_h = 0;
    asm volatile("" : "+s"(ofs_w));
    asm volatile("" : "+s"(ofs_h));
    const __bf16* wih = wih0 + ofs_w;
    const __bf16* whh = whh0 + ofs_h;

    f32x8 acc[8] = {};                      // [ct*4 + gate]
#pragma unroll
    for (int kt = 0; kt < 8; ++kt) {        // K over E (x @ W_ih^T)
      bf16x16 ax = load_a_frag(arow_x, kt << 5, base);
#pragma unroll
      for (int ct = 0; ct < 2; ++ct) {
#pragma unroll
        for (int g = 0; g < 4; ++g) {
          int nt = (g << 4) + (wave << 1) + ct;
          const __bf16* bp = wih + (((size_t)(nt << 3) + kt) << 9) + (lane << 4);
          acc[ct * 4 + g] = wmma_bf16(ax, *(const bf16x16*)bp, acc[ct * 4 + g]);
        }
      }
    }
#pragma unroll
    for (int kt = 0; kt < 8; ++kt) {        // K over H (h @ W_hh^T)
      bf16x16 ah = load_a_frag(arow_h, kt << 5, base);
#pragma unroll
      for (int ct = 0; ct < 2; ++ct) {
#pragma unroll
        for (int g = 0; g < 4; ++g) {
          int nt = (g << 4) + (wave << 1) + ct;
          const __bf16* bp = whh + (((size_t)(nt << 3) + kt) << 9) + (lane << 4);
          acc[ct * 4 + g] = wmma_bf16(ah, *(const bf16x16*)bp, acc[ct * 4 + g]);
        }
      }
    }
    __syncthreads();                        // all h_lds reads done before rewrite

#pragma unroll
    for (int ct = 0; ct < 2; ++ct) {
      int col = (wave << 5) + (ct << 4) + nlo;
#pragma unroll
      for (int r = 0; r < 8; ++r) {
        float gi = sigf(acc[ct * 4 + 0][r] + bI[ct]);
        float gf = sigf(acc[ct * 4 + 1][r] + bF[ct]);
        float gg = tanhf(acc[ct * 4 + 2][r] + bG[ct]);
        float go = sigf(acc[ct * 4 + 3][r] + bO[ct]);
        float c = gf * c_reg[ct][r] + gi * gg;
        c_reg[ct][r] = c;
        float h = go * tanhf(c);
        int m = r + mrow;
        __bf16 hb = (__bf16)h;
        h_lds[m * XP + col] = hb;           // feeds next step's A fragment
        h_all[(size_t)(s * BATCH + b0 + m) * HD2 + (dir ? HID : 0) + col] = hb;
      }
    }
  }
}

// K3: emissions = [h_f|h_b] @ W_out^T + b_out, padded to 64 cols.
__global__ void __launch_bounds__(256) k_emis(const __bf16* __restrict__ h_all,
                                              const __bf16* __restrict__ wout,
                                              const float* __restrict__ b_out,
                                              float* __restrict__ emis) {
  int tid = threadIdx.x, wave = tid >> 5, lane = tid & 31;
  int rt = blockIdx.x * 8 + wave;
  int r0 = rt << 4;
  int nlo = lane & 15, base = (lane >> 4) << 3, mrow = (lane >> 4) << 3;
  const __bf16* arow = h_all + (size_t)(r0 + nlo) * HD2;
  __builtin_prefetch(arow + (size_t)16 * HD2);   // next row-tile of activations
  f32x8 acc[4] = {};
#pragma unroll
  for (int kt = 0; kt < 16; ++kt) {
    bf16x16 a = load_a_frag(arow, kt << 5, base);
#pragma unroll
    for (int nt = 0; nt < 4; ++nt) {
      const __bf16* bp = wout + (((size_t)(nt << 4) + kt) << 9) + (lane << 4);
      acc[nt] = wmma_bf16(a, *(const bf16x16*)bp, acc[nt]);
    }
  }
#pragma unroll
  for (int nt = 0; nt < 4; ++nt) {
    int col = (nt << 4) + nlo;
    float bo = (col < TAGS) ? b_out[col] : 0.0f;
#pragma unroll
    for (int r = 0; r < 8; ++r) {
      int row = r0 + r + mrow;
      emis[(size_t)row * TPAD + col] = (col < TAGS) ? (acc[nt][r] + bo) : -1e30f;
    }
  }
}

// K4: gold-path score (one thread per batch element).
__global__ void k_crf_num(const int* __restrict__ tags, const uint8_t* __restrict__ mask,
                          const float* __restrict__ emis,
                          const float* __restrict__ start_trans,
                          const float* __restrict__ end_trans,
                          const float* __restrict__ trans, float* __restrict__ numv) {
  int b = threadIdx.x;
  if (b >= BATCH) return;
  int prev = tags[b * SLEN];
  float num = start_trans[prev] + emis[(size_t)b * TPAD + prev];
  int msum = mask[b * SLEN] ? 1 : 0;
  for (int s = 1; s < SLEN; ++s) {
    int cur = tags[b * SLEN + s];
    float m = mask[b * SLEN + s] ? 1.0f : 0.0f;
    num += m * (trans[prev * TAGS + cur] + emis[((size_t)s * BATCH + b) * TPAD + cur]);
    msum += mask[b * SLEN + s] ? 1 : 0;
    prev = cur;
  }
  num += end_trans[tags[b * SLEN + (msum - 1)]];
  numv[b] = num;
}

// K5: log-partition forward scan; one block per batch element, trans in LDS.
__global__ void __launch_bounds__(64) k_crf_fwd(const uint8_t* __restrict__ mask,
                                                const float* __restrict__ emis,
                                                const float* __restrict__ start_trans,
                                                const float* __restrict__ end_trans,
                                                const float* __restrict__ trans,
                                                float* __restrict__ logZ) {
  __shared__ float trS[TAGS * TAGS];
  __shared__ float alpha[TAGS];
  int b = blockIdx.x, j = threadIdx.x;
  for (int i = j; i < TAGS * TAGS; i += 64) trS[i] = trans[i];
  if (j < TAGS) alpha[j] = start_trans[j] + emis[(size_t)b * TPAD + j];
  __syncthreads();
  for (int s = 1; s < SLEN; ++s) {
    float nxt = 0.0f;
    if (j < TAGS) {
      float e = emis[((size_t)s * BATCH + b) * TPAD + j];
      float mx = -1e30f;
      for (int i = 0; i < TAGS; ++i) mx = fmaxf(mx, alpha[i] + trS[i * TAGS + j]);
      float sum = 0.0f;
      for (int i = 0; i < TAGS; ++i) sum += __expf(alpha[i] + trS[i * TAGS + j] - mx);
      nxt = mx + __logf(sum) + e;
      if (!mask[b * SLEN + s]) nxt = alpha[j];
    }
    __syncthreads();
    if (j < TAGS) alpha[j] = nxt;
    __syncthreads();
  }
  if (j == 0) {
    float mx = -1e30f;
    for (int i = 0; i < TAGS; ++i) mx = fmaxf(mx, alpha[i] + end_trans[i]);
    float sum = 0.0f;
    for (int i = 0; i < TAGS; ++i) sum += __expf(alpha[i] + end_trans[i] - mx);
    logZ[b] = mx + __logf(sum);
  }
}

// K6: loss = mean(logZ - num)
__global__ void k_reduce(const float* __restrict__ numv, const float* __restrict__ logZ,
                         float* __restrict__ out) {
  __shared__ float sh[BATCH];
  int t = threadIdx.x;
  sh[t] = logZ[t] - numv[t];
  __syncthreads();
  for (int off = 64; off > 0; off >>= 1) {
    if (t < off) sh[t] += sh[t + off];
    __syncthreads();
  }
  if (t == 0) out[0] = sh[0] / (float)BATCH;
}

extern "C" void kernel_launch(void* const* d_in, const int* in_sizes, int n_in,
                              void* d_out, int out_size, void* d_ws, size_t ws_size,
                              hipStream_t stream) {
  const int*     sentences = (const int*)d_in[0];
  const int*     tags      = (const int*)d_in[1];
  const uint8_t* mask      = (const uint8_t*)d_in[2];
  const float*   embedding = (const float*)d_in[3];
  const float*   w_ih_f = (const float*)d_in[4];
  const float*   w_hh_f = (const float*)d_in[5];
  const float*   b_ih_f = (const float*)d_in[6];
  const float*   b_hh_f = (const float*)d_in[7];
  const float*   w_ih_b = (const float*)d_in[8];
  const float*   w_hh_b = (const float*)d_in[9];
  const float*   b_ih_b = (const float*)d_in[10];
  const float*   b_hh_b = (const float*)d_in[11];
  const float*   W_out  = (const float*)d_in[12];
  const float*   b_out  = (const float*)d_in[13];
  const float*   start_trans = (const float*)d_in[14];
  const float*   end_trans   = (const float*)d_in[15];
  const float*   trans       = (const float*)d_in[16];
  (void)in_sizes; (void)n_in; (void)out_size; (void)ws_size;

  char* basep = (char*)d_ws;
  size_t off = 0;
  auto alloc = [&](size_t bytes) -> char* {
    off = (off + 255) & ~(size_t)255;
    char* p = basep + off;
    off += bytes;
    return p;
  };
  __bf16* xs    = (__bf16*)alloc((size_t)SLEN * BATCH * EMB * 2);
  __bf16* h_all = (__bf16*)alloc((size_t)SLEN * BATCH * HD2 * 2);
  float*  emis  = (float*) alloc((size_t)SLEN * BATCH * TPAD * 4);
  __bf16* wihfS = (__bf16*)alloc((size_t)GATES * EMB * 2);
  __bf16* wihbS = (__bf16*)alloc((size_t)GATES * EMB * 2);
  __bf16* whhfS = (__bf16*)alloc((size_t)GATES * HID * 2);
  __bf16* whhbS = (__bf16*)alloc((size_t)GATES * HID * 2);
  __bf16* woutS = (__bf16*)alloc((size_t)HD2 * TPAD * 2);
  float*  biasF = (float*) alloc(GATES * 4);
  float*  biasB = (float*) alloc(GATES * 4);
  float*  numv  = (float*) alloc(BATCH * 4);
  float*  logZ  = (float*) alloc(BATCH * 4);

  k_swizzle_w<<<(GATES * EMB + 255) / 256, 256, 0, stream>>>(w_ih_f, GATES, GATES, EMB, wihfS);
  k_swizzle_w<<<(GATES * EMB + 255) / 256, 256, 0, stream>>>(w_ih_b, GATES, GATES, EMB, wihbS);
  k_swizzle_w<<<(GATES * HID + 255) / 256, 256, 0, stream>>>(w_hh_f, GATES, GATES, HID, whhfS);
  k_swizzle_w<<<(GATES * HID + 255) / 256, 256, 0, stream>>>(w_hh_b, GATES, GATES, HID, whhbS);
  k_swizzle_w<<<(TPAD * HD2 + 255) / 256, 256, 0, stream>>>(W_out, TAGS, TPAD, HD2, woutS);
  k_bias<<<(2 * GATES + 255) / 256, 256, 0, stream>>>(b_ih_f, b_hh_f, b_ih_b, b_hh_b, biasF, biasB);
  k_embed<<<SLEN * BATCH, EMB, 0, stream>>>(sentences, embedding, xs);
  k_lstm<<<16, 256, 0, stream>>>(xs, wihfS, wihbS, whhfS, whhbS, biasF, biasB, h_all);
  k_emis<<<(SLEN * BATCH / 16) / 8, 256, 0, stream>>>(h_all, woutS, b_out, emis);
  k_crf_num<<<1, BATCH, 0, stream>>>(tags, mask, emis, start_trans, end_trans, trans, numv);
  k_crf_fwd<<<BATCH, 64, 0, stream>>>(mask, emis, start_trans, end_trans, trans, logZ);
  k_reduce<<<1, BATCH, 0, stream>>>(numv, logZ, (float*)d_out);
}
